// KGIN_26130581028994
// MI455X (gfx1250) — matrix operand order repaired and checked by reading, not compiled, not run on hardware
//
#include <hip/hip_runtime.h>
#include <hip/hip_bf16.h>

// KGIN-style relational GNN for MI455X (gfx1250, wave32, WMMA).
// Heavy per-edge type-indexed matmuls (128->64, 128->128) run on
// v_wmma_f32_16x16x32_f16 over 16-edge tiles that are pre-sorted by edge
// type (counting sort, segments padded to multiples of 16 with -1).
// Embedding gathers stay in the 192MB L2 (25.6MB table); segment sums use
// global f32 atomics (order nondeterministic in rounding, as any
// atomic-based segment_sum is).

#define DIM     128
#define RDIM    64
#define IDIM    8
#define NET     9
#define NNT     3
#define NLAYERS 3

typedef __attribute__((ext_vector_type(16))) _Float16 v16h;
typedef __attribute__((ext_vector_type(8)))  _Float16 v8h;
typedef __attribute__((ext_vector_type(8)))  float    v8f;

// ---- A-operand loader: 16-bit A 16x32 layout (ISA 7.12.2) ----------------
// lane l holds row (l&15); K halves {kb..kb+7} in h[0..7] and
// {kb+16..kb+23} in h[8..15], kb = (l<16 ? 0 : 8) relative to the K-chunk.
static __device__ __forceinline__ v16h load_a16(const _Float16* p) {
  v8h lo = *(const v8h*)(p);
  v8h hi = *(const v8h*)(p + 16);
  v16h a;
#pragma unroll
  for (int i = 0; i < 8; ++i) { a[i] = lo[i]; a[8 + i] = hi[i]; }
  return a;
}

static __device__ __forceinline__ v16h zero_a16() {
  v16h a;
#pragma unroll
  for (int i = 0; i < 16; ++i) a[i] = (_Float16)0.f;
  return a;
}

// ---- counting-sort of edges by edge type ---------------------------------
__global__ void k_count(const int* __restrict__ et, int E, int* __restrict__ counts) {
  int i = blockIdx.x * blockDim.x + threadIdx.x;
  if (i < E) atomicAdd(&counts[et[i]], 1);
}

__global__ void k_offsets(const int* __restrict__ counts, int* __restrict__ cursor) {
  if (blockIdx.x == 0 && threadIdx.x == 0) {
    int acc = 0;
    for (int t = 0; t < NET; ++t) {
      cursor[t] = acc;                       // 16-aligned segment start
      acc += (counts[t] + 15) & ~15;
    }
  }
}

__global__ void k_scatter(const int* __restrict__ et, int E,
                          int* __restrict__ cursor, int* __restrict__ perm) {
  int i = blockIdx.x * blockDim.x + threadIdx.x;
  if (i < E) {
    int p = atomicAdd(&cursor[et[i]], 1);
    perm[p] = i;
  }
}

// ---- weight convert: W[t][k][n] f32 -> WT[t][n][k] f16 -------------------
__global__ void k_wconvT(const float* __restrict__ W, _Float16* __restrict__ WT,
                         int T, int K, int Nc) {
  long total = (long)T * K * Nc;
  long i = (long)blockIdx.x * blockDim.x + threadIdx.x;
  if (i < total) {
    int t   = (int)(i / ((long)K * Nc));
    int rem = (int)(i - (long)t * K * Nc);
    int k = rem / Nc, n = rem % Nc;
    WT[((long)t * Nc + n) * K + k] = (_Float16)W[i];
  }
}

__global__ void k_f32_to_f16(const float* __restrict__ x, _Float16* __restrict__ y, long n) {
  long i = (long)blockIdx.x * blockDim.x + threadIdx.x;
  long stride = (long)gridDim.x * blockDim.x;
  for (; i < n; i += stride) y[i] = (_Float16)x[i];
}

// ---- score kernel: one wave32 per 16-edge tile (uniform edge type) -------
__global__ void k_score(const int* __restrict__ perm,
                        const int* __restrict__ sources,
                        const int* __restrict__ targets,
                        const int* __restrict__ ttypes,
                        const int* __restrict__ etypes,
                        const float* __restrict__ embf,
                        const _Float16* __restrict__ emb16,
                        const float* __restrict__ W_intent,
                        const float* __restrict__ b_intent,
                        const float* __restrict__ W_ibn,
                        const float* __restrict__ b_ibn,
                        const _Float16* __restrict__ WsrcT,
                        const float* __restrict__ b_src,
                        const float* __restrict__ et_emb,
                        float* __restrict__ exp_s,
                        float* __restrict__ denom) {
  __shared__ float s_iw[16];
  __shared__ float s_red[16][17];
  const int  lane = threadIdx.x;
  const long base = (long)blockIdx.x * 16;
  const int  e0   = perm[base];
  if (e0 < 0) return;                         // fully-dead tile (uniform)
  const int t = etypes[e0];                   // whole tile shares edge type

  // --- intent weight (VALU path, lanes 0..15, edge m = lane) ---
  if (lane < 16) {
    const int e = perm[base + lane];
    float w = 0.f;
    if (e >= 0) {
      const int    tt   = ttypes[e];
      const float* trow = embf + (long)targets[e] * DIM;
      const float* WI   = W_intent + (long)tt * DIM * IDIM;
      float it[IDIM];
#pragma unroll
      for (int i = 0; i < IDIM; ++i) it[i] = b_intent[tt * IDIM + i];
      for (int k = 0; k < DIM; k += 4) {
        const float4 xv = *(const float4*)(trow + k);
        const float  xs[4] = {xv.x, xv.y, xv.z, xv.w};
#pragma unroll
        for (int j = 0; j < 4; ++j) {
          const float4 w0 = *(const float4*)(WI + (k + j) * IDIM);
          const float4 w1 = *(const float4*)(WI + (k + j) * IDIM + 4);
          it[0] += xs[j] * w0.x; it[1] += xs[j] * w0.y;
          it[2] += xs[j] * w0.z; it[3] += xs[j] * w0.w;
          it[4] += xs[j] * w1.x; it[5] += xs[j] * w1.y;
          it[6] += xs[j] * w1.z; it[7] += xs[j] * w1.w;
        }
      }
      float mx = it[0];
#pragma unroll
      for (int i = 1; i < IDIM; ++i) mx = fmaxf(mx, it[i]);
      float s = 0.f;
#pragma unroll
      for (int i = 0; i < IDIM; ++i) { it[i] = expf(it[i] - mx); s += it[i]; }
      const float inv = 1.f / s;
#pragma unroll
      for (int i = 0; i < IDIM; ++i) it[i] *= inv;

      const float* WB = W_ibn + (long)tt * IDIM * NET;
      float z[NET];
#pragma unroll
      for (int j = 0; j < NET; ++j) z[j] = b_ibn[tt * NET + j];
#pragma unroll
      for (int i = 0; i < IDIM; ++i) {
        const float x = it[i];
#pragma unroll
        for (int j = 0; j < NET; ++j) z[j] += x * WB[i * NET + j];
      }
      float mz = z[0];
#pragma unroll
      for (int j = 1; j < NET; ++j) mz = fmaxf(mz, z[j]);
      float sz = 0.f, ze[NET];
#pragma unroll
      for (int j = 0; j < NET; ++j) { ze[j] = expf(z[j] - mz); sz += ze[j]; }
      w = ze[etypes[e]] / sz;
    }
    s_iw[lane] = w;
  }

  // --- A tile: f16 source-embedding rows ---
  const int arow = lane & 15;
  const int ea   = perm[base + arow];
  const int kbA  = (lane < 16) ? 0 : 8;
  v16h A[4];
  if (ea >= 0) {
    const _Float16* rb = emb16 + (long)sources[ea] * DIM;
#pragma unroll
    for (int c = 0; c < 4; ++c) A[c] = load_a16(rb + c * 32 + kbA);
  } else {
#pragma unroll
    for (int c = 0; c < 4; ++c) A[c] = zero_a16();
  }

  // --- src_rel = selu(src_e @ W_src[t] + b_src[t]); partial dot with et_emb
  const int ncol = lane & 15;
  const int kbB  = (lane < 16) ? 0 : 8;
  float partial[8];
#pragma unroll
  for (int v = 0; v < 8; ++v) partial[v] = 0.f;
  const float sscale = 1.0507009873554805f;
  const float salpha = 1.6732632423543773f;
  for (int nt = 0; nt < RDIM / 16; ++nt) {
    const int col = nt * 16 + ncol;
    const _Float16* cb = WsrcT + ((long)t * RDIM + col) * DIM;
    v8f acc;
#pragma unroll
    for (int v = 0; v < 8; ++v) acc[v] = 0.f;
#pragma unroll
    for (int c = 0; c < 4; ++c) {
      const v16h b = load_a16(cb + c * 32 + kbB);
      acc = __builtin_amdgcn_wmma_f32_16x16x32_f16(false, A[c], false, b,
                                                   (short)0, acc, false, false);
    }
    const float bb = b_src[t * RDIM + col];
    const float ee = et_emb[t * RDIM + col];
#pragma unroll
    for (int v = 0; v < 8; ++v) {
      const float x  = acc[v] + bb;
      const float sl = (x > 0.f) ? sscale * x : sscale * salpha * expm1f(x);
      partial[v] += sl * ee;
    }
  }
  // C/D layout: lane<16 holds rows 0..7 (VGPR v -> row v), lane>=16 rows 8..15
  const int rbase = (lane < 16) ? 0 : 8;
#pragma unroll
  for (int v = 0; v < 8; ++v) s_red[rbase + v][ncol] = partial[v];
  __syncthreads();

  if (lane < 16) {
    const int e = perm[base + lane];
    if (e >= 0) {
      float s = 0.f;
#pragma unroll
      for (int c = 0; c < 16; ++c) s += s_red[lane][c];
      const float es = expf(s * s_iw[lane]);
      exp_s[e] = es;
      atomicAdd(&denom[targets[e]], es);
    }
  }
}

// ---- edge-softmax normalize (reciprocal_no_nan) --------------------------
__global__ void k_dist(const float* __restrict__ exp_s, const int* __restrict__ targets,
                       const float* __restrict__ denom, float* __restrict__ dist_out,
                       int E) {
  int i = blockIdx.x * blockDim.x + threadIdx.x;
  if (i < E) {
    const float d = denom[targets[i]];
    dist_out[i] = (d == 0.f) ? 0.f : exp_s[i] / d;
  }
}

// ---- message kernel: 16x128 WMMA tile per 16 edges, scatter-add ----------
__global__ void k_msg(const int* __restrict__ perm,
                      const int* __restrict__ sources,
                      const int* __restrict__ targets,
                      const int* __restrict__ etypes,
                      const _Float16* __restrict__ emb16,
                      const _Float16* __restrict__ WmsgT,
                      const float* __restrict__ dist,
                      float* __restrict__ emb_out) {
  const int  lane = threadIdx.x;
  const long base = (long)blockIdx.x * 16;
  const int  e0   = perm[base];
  if (e0 < 0) return;
  const int t = etypes[e0];

  const int arow = lane & 15;
  const int ea   = perm[base + arow];
  const int kbA  = (lane < 16) ? 0 : 8;
  v16h A[4];
  if (ea >= 0) {
    const _Float16* rb = emb16 + (long)sources[ea] * DIM;
#pragma unroll
    for (int c = 0; c < 4; ++c) A[c] = load_a16(rb + c * 32 + kbA);
  } else {
#pragma unroll
    for (int c = 0; c < 4; ++c) A[c] = zero_a16();
  }

  const int rbase = (lane < 16) ? 0 : 8;
  float dv[8];
  int   tg[8];
#pragma unroll
  for (int v = 0; v < 8; ++v) {
    const int e = perm[base + rbase + v];
    if (e >= 0) { dv[v] = dist[e]; tg[v] = targets[e]; }
    else        { dv[v] = 0.f;     tg[v] = -1; }
  }

  const int ncol = lane & 15;
  const int kbB  = (lane < 16) ? 0 : 8;
  for (int nt = 0; nt < DIM / 16; ++nt) {
    const int col = nt * 16 + ncol;
    const _Float16* cb = WmsgT + ((long)t * DIM + col) * DIM;
    v8f acc;
#pragma unroll
    for (int v = 0; v < 8; ++v) acc[v] = 0.f;
#pragma unroll
    for (int c = 0; c < 4; ++c) {
      const v16h b = load_a16(cb + c * 32 + kbB);
      acc = __builtin_amdgcn_wmma_f32_16x16x32_f16(false, A[c], false, b,
                                                   (short)0, acc, false, false);
    }
#pragma unroll
    for (int v = 0; v < 8; ++v) {
      if (tg[v] >= 0)
        atomicAdd(emb_out + (long)tg[v] * DIM + col, acc[v] * dv[v]);
    }
  }
}

extern "C" void kernel_launch(void* const* d_in, const int* in_sizes, int n_in,
                              void* d_out, int out_size, void* d_ws, size_t ws_size,
                              hipStream_t stream) {
  const float* entity   = (const float*)d_in[0];
  const int*   sources  = (const int*)d_in[1];
  const int*   targets  = (const int*)d_in[2];
  const int*   ttypes   = (const int*)d_in[3];
  const int*   etypes   = (const int*)d_in[4];
  const float* et_emb   = (const float*)d_in[5];
  const float* W_src    = (const float*)d_in[6];
  const float* b_src    = (const float*)d_in[7];
  const float* W_msg    = (const float*)d_in[8];
  const float* W_intent = (const float*)d_in[9];
  const float* b_intent = (const float*)d_in[10];
  const float* W_ibn    = (const float*)d_in[11];
  const float* b_ibn    = (const float*)d_in[12];
  (void)n_in; (void)out_size; (void)ws_size;

  const int N        = in_sizes[0] / DIM;
  const int E        = in_sizes[1];
  const int numTiles = (E + 15) / 16 + NET;   // >= padded sorted length / 16
  const int permLen  = numTiles * 16;

  char* w = (char*)d_ws;
  auto take = [&](size_t bytes) {
    char* p = w;
    w += (bytes + 255) & ~(size_t)255;
    return p;
  };
  float*    wsA    = (float*)take((size_t)N * DIM * 4);
  float*    wsB    = (float*)take((size_t)N * DIM * 4);
  _Float16* emb16  = (_Float16*)take((size_t)N * DIM * 2);
  _Float16* wsrcT  = (_Float16*)take((size_t)NET * RDIM * DIM * 2);
  _Float16* wmsgT  = (_Float16*)take((size_t)NET * DIM * DIM * 2);
  float*    exps   = (float*)take((size_t)E * 4);
  float*    denom  = (float*)take((size_t)N * 4);
  int*      perm   = (int*)take((size_t)permLen * 4);
  int*      counts = (int*)take(64);
  int*      cursor = (int*)take(64);

  // --- one-time per call: sort edges by type, convert weights to f16^T ---
  hipMemsetAsync(counts, 0, 64, stream);
  k_count<<<(E + 255) / 256, 256, 0, stream>>>(etypes, E, counts);
  k_offsets<<<1, 1, 0, stream>>>(counts, cursor);
  hipMemsetAsync(perm, 0xFF, (size_t)permLen * 4, stream);   // -1 padding
  k_scatter<<<(E + 255) / 256, 256, 0, stream>>>(etypes, E, cursor, perm);
  {
    long tot = (long)NET * DIM * RDIM;
    k_wconvT<<<(int)((tot + 255) / 256), 256, 0, stream>>>(W_src, wsrcT, NET, DIM, RDIM);
    tot = (long)NET * DIM * DIM;
    k_wconvT<<<(int)((tot + 255) / 256), 256, 0, stream>>>(W_msg, wmsgT, NET, DIM, DIM);
  }

  float* out_emb  = (float*)d_out;
  float* out_dist = (float*)d_out + (size_t)N * DIM;

  for (int layer = 0; layer < NLAYERS; ++layer) {
    const float* ein  = (layer == 0) ? entity : ((layer == 1) ? wsA : wsB);
    float*       eout = (layer == 0) ? wsA    : ((layer == 1) ? wsB : out_emb);
    float*       dist = out_dist + (size_t)layer * E;

    k_f32_to_f16<<<2048, 256, 0, stream>>>(ein, emb16, (long)N * DIM);
    hipMemsetAsync(denom, 0, (size_t)N * 4, stream);
    hipMemsetAsync(eout, 0, (size_t)N * DIM * 4, stream);

    k_score<<<numTiles, 32, 0, stream>>>(perm, sources, targets, ttypes, etypes,
                                         ein, emb16, W_intent, b_intent, W_ibn, b_ibn,
                                         wsrcT, b_src, et_emb, exps, denom);
    k_dist<<<(E + 255) / 256, 256, 0, stream>>>(exps, targets, denom, dist, E);
    k_msg<<<numTiles, 32, 0, stream>>>(perm, sources, targets, etypes,
                                       emb16, wmsgT, dist, eout);
  }
}
